// PathEmbedding_3874060501716
// MI455X (gfx1250) — matrix operand order repaired and checked by reading, not compiled, not run on hardware
//
#include <hip/hip_runtime.h>

#define B_    16
#define E_    1024
#define DIN_  128
#define P_    2048
#define D_    128
#define L_    16
#define TD_   384          // 3*D
#define NSEQ_ (B_*P_)      // 32768
#define NG_   (B_*E_)      // 16384 distinct gathered rows

typedef __attribute__((ext_vector_type(16))) _Float16 v16h;
typedef __attribute__((ext_vector_type(8)))  _Float16 v8h;
typedef __attribute__((ext_vector_type(8)))  float    v8f;
typedef __attribute__((ext_vector_type(4))) unsigned int u32x4;
typedef __attribute__((ext_vector_type(8))) int         i32x8;
typedef __attribute__((ext_vector_type(4))) int         i32x4;

#if __has_builtin(__builtin_amdgcn_tensor_load_to_lds)
#define HAVE_TDM 1
#else
#define HAVE_TDM 0
#endif

__device__ __forceinline__ float sigmoidf_(float x) { return 1.0f / (1.0f + __expf(-x)); }
__device__ __forceinline__ float tanhf_(float x)    { return 2.0f * sigmoidf_(2.0f * x) - 1.0f; }

// D = A*B + C,  16x16x32 f16 -> f32
__device__ __forceinline__ v8f wmma_(v16h a, v16h b, v8f c) {
    return __builtin_amdgcn_wmma_f32_16x16x32_f16(false, a, false, b, (short)0, c, false, false);
}

// Build an A-fragment (16x32 f16) from a row of 128 contiguous f16.
// Lane layout: lanes hsel=0 hold K = kc*32 + {0..15},
// lanes hsel=1 hold K = kc*32 + {16..31}; row M = lane%16.
__device__ __forceinline__ v16h load_afrag(const _Float16* row, int kc, int hsel) {
    return *(const v16h*)(row + kc * 32 + hsel * 16);
}

// --------------------------------------------------------------- TDM stage
// DMA 24576 dwords (W_rec^T, 96 KB f16) from global into LDS offset 0 as a
// single-row 2D tile.  D# packing per CDNA5 ISA.
#if HAVE_TDM
__device__ __forceinline__ void tdm_load_wrec(const _Float16* gsrc) {
    unsigned long long ga = (unsigned long long)(size_t)(const void*)gsrc;
    u32x4 g0;
    g0[0] = 1u;                                   // count=1 (valid user D#)
    g0[1] = 0u;                                   // lds_addr = 0
    g0[2] = (unsigned)(ga & 0xffffffffu);         // global_addr[31:0]
    g0[3] = (unsigned)((ga >> 32) & 0x1ffffffu)   // global_addr[56:32]
          | (2u << 30);                           // type = 2 ("image")
    i32x8 g1;
    g1[0] = 0x20000;                              // data_size=2 (4 bytes)
    g1[1] = (int)(24576u << 16);                  // tensor_dim0[15:0]  @[63:48]
    g1[2] = (int)(1u << 16);                      // dim0 hi=0 | tensor_dim1=1
    g1[3] = (int)(24576u << 16);                  // dim1 hi=0 | tile_dim0=24576
    g1[4] = 1;                                    // tile_dim1=1, tile_dim2=0
    g1[5] = 24576;                                // tensor_dim0_stride[31:0]
    g1[6] = (int)(24576u << 16);                  // stride0 hi=0 | stride1 lo16
    g1[7] = 0;                                    // stride1 hi
    i32x4 z4 = {0, 0, 0, 0};
#if __clang_major__ >= 23
    i32x8 z8 = {0, 0, 0, 0, 0, 0, 0, 0};
    __builtin_amdgcn_tensor_load_to_lds(g0, g1, z4, z4, z8, 0);
#else
    __builtin_amdgcn_tensor_load_to_lds(g0, g1, z4, z4, 0);
#endif
#if __has_builtin(__builtin_amdgcn_s_wait_tensorcnt)
    __builtin_amdgcn_s_wait_tensorcnt(0);
#endif
}
#endif

// ---------------------------------------------------------------- topology
__global__ void k_init_elem(int* elemtab) {
    int id = blockIdx.x * blockDim.x + threadIdx.x;
    if (id < P_ * L_) elemtab[id] = -1;
}
__global__ void k_scatter(const int* paths, const int* idx, const int* seqs,
                          int ntot, int* elemtab) {
    int k = blockIdx.x * blockDim.x + threadIdx.x;
    if (k < ntot) elemtab[idx[k] * L_ + seqs[k]] = paths[k];
}

// --------------------------------------------- f16 col-major weight copies
__global__ void k_prep(const float* W_in, const float* W_rec, const float* wv,
                       _Float16* WinT, _Float16* WrecT, _Float16* WvT) {
    int id = blockIdx.x * blockDim.x + threadIdx.x;
    if (id < TD_ * DIN_) {
        int c = id >> 7, k = id & 127;
        WinT[c * 128 + k]  = (_Float16)W_in[k * TD_ + c];
        WrecT[c * 128 + k] = (_Float16)W_rec[k * TD_ + c];
    }
    if (id < D_ * D_) {
        int d = id >> 7, j = id & 127;
        WvT[d * 128 + j] = (_Float16)wv[j * 128 + d];
    }
}

// Tt[j*128+i] = T[i,j] = sum_d wq[i,d]*wk[j,d]   (so Tt is a ready B-operand)
__global__ void k_T(const float* wq, const float* wk, _Float16* Tt) {
    int id = blockIdx.x * blockDim.x + threadIdx.x;
    if (id >= D_ * D_) return;
    int j = id >> 7, i = id & 127;
    float s = 0.f;
    for (int d = 0; d < 128; ++d) s += wq[i * 128 + d] * wk[j * 128 + d];
    Tt[j * 128 + i] = (_Float16)s;
}

// ---------------------------------------- G = inputs_flat @ W_in + b_in (f32)
__global__ void k_G(const float* inputs, const _Float16* WinT, const float* b_in,
                    float* G) {
    int wid  = (blockIdx.x * blockDim.x + threadIdx.x) >> 5;
    int lane = threadIdx.x & 31;
    int rt = wid / 6, nbg = wid % 6;     // 1024 row tiles x 6 col groups
    int hsel = lane >> 4, ncol = lane & 15;
    const float* arow = inputs + (size_t)(rt * 16 + (lane & 15)) * DIN_;
    v16h af[4];
#pragma unroll
    for (int kc = 0; kc < 4; ++kc) {
        v16h a;
#pragma unroll
        for (int e = 0; e < 8; ++e) {
            a[e]     = (_Float16)arow[kc * 32 + hsel * 16 + e];
            a[8 + e] = (_Float16)arow[kc * 32 + hsel * 16 + 8 + e];
        }
        af[kc] = a;
    }
#pragma unroll
    for (int nbl = 0; nbl < 4; ++nbl) {
        int nb = nbg * 4 + nbl;
        v8f acc = {};
#pragma unroll
        for (int kc = 0; kc < 4; ++kc) {
            v16h b = *(const v16h*)(WinT + (nb * 16 + ncol) * 128 + kc * 32 + hsel * 16);
            acc = wmma_(af[kc], b, acc);
        }
        float bi = b_in[nb * 16 + ncol];
#pragma unroll
        for (int r = 0; r < 8; ++r)
            G[(size_t)(rt * 16 + hsel * 8 + r) * TD_ + nb * 16 + ncol] = acc[r] + bi;
    }
}

// ---------------------------------------------------- fused 16-step GRU scan
// 4 waves/block, one 16-seq tile per wave.  Dynamic LDS:
//   [0, 98304)            WrecT staged (384 cols x 128 K, f16)  <- TDM DMA
//   [98304 + wave*4096)   per-wave h stage: 16 rows x 128 f16
__global__ void k_gru(const float* G, const _Float16* WrecT,
                      const float* b_in, const float* b_rec,
                      const int* elemtab, _Float16* HS) {
    extern __shared__ char smem[];
    _Float16* WrecL = (_Float16*)smem;
    int widx = threadIdx.x >> 5;
    int lane = threadIdx.x & 31;
    _Float16* hst = (_Float16*)(smem + 98304 + widx * 4096);

#if HAVE_TDM
    if (widx == 0) tdm_load_wrec(WrecT);   // one TDM DMA per workgroup
#else
    {   // fallback: cooperative stage of W_rec^T (96 KB) into LDS
        const uint4* src = (const uint4*)WrecT;
        uint4* dst = (uint4*)WrecL;
        for (int i = threadIdx.x; i < 6144; i += 128) dst[i] = src[i];
    }
#endif
    {   // h0 = 0 in the per-wave stage
        uint4 z = make_uint4(0u, 0u, 0u, 0u);
        uint4* hz = (uint4*)hst;
        for (int i = lane; i < 256; i += 32) hz[i] = z;
    }
    __syncthreads();

    int tile = blockIdx.x * 4 + widx;
    int s0 = tile * 16;
    int bb = s0 >> 11;            // batch index (tile never crosses b)
    int p0 = s0 & (P_ - 1);
    int hsel = lane >> 4, ncol = lane & 15;

    float h[8][8];
#pragma unroll
    for (int db = 0; db < 8; ++db)
#pragma unroll
        for (int r = 0; r < 8; ++r) h[db][r] = 0.f;

    for (int t = 0; t < L_; ++t) {
        // A fragments of current h (16 x 128) from LDS stage
        v16h af[4];
#pragma unroll
        for (int kc = 0; kc < 4; ++kc)
            af[kc] = load_afrag(hst + (lane & 15) * 128, kc, hsel);

        // per-row gather offsets into G (or -1 => padded step, use b_in)
        int gr[8];
#pragma unroll
        for (int r = 0; r < 8; ++r) {
            int m = hsel * 8 + r;
            int e = elemtab[(p0 + m) * L_ + t];
            gr[r] = (e < 0) ? -1 : (bb * E_ + e) * TD_;
        }

#pragma unroll
        for (int db = 0; db < 8; ++db) {
            v8f cz = {}, cr = {}, ch = {};
#pragma unroll
            for (int kc = 0; kc < 4; ++kc) {
                v16h bz = *(const v16h*)(WrecL + ((db     ) * 16 + ncol) * 128 + kc * 32 + hsel * 16);
                v16h br = *(const v16h*)(WrecL + ((db +  8) * 16 + ncol) * 128 + kc * 32 + hsel * 16);
                v16h bh = *(const v16h*)(WrecL + ((db + 16) * 16 + ncol) * 128 + kc * 32 + hsel * 16);
                cz = wmma_(af[kc], bz, cz);
                cr = wmma_(af[kc], br, cr);
                ch = wmma_(af[kc], bh, ch);
            }
            int c0 = db * 16 + ncol;
            float brz = b_rec[c0], brr = b_rec[128 + c0], brh = b_rec[256 + c0];
            float biz = b_in[c0],  bir = b_in[128 + c0],  bih = b_in[256 + c0];
#pragma unroll
            for (int r = 0; r < 8; ++r) {
                float xz, xr, xh;
                if (gr[r] >= 0) {
                    const float* g = G + gr[r];
                    xz = g[c0]; xr = g[128 + c0]; xh = g[256 + c0];
                } else { xz = biz; xr = bir; xh = bih; }
                float z  = sigmoidf_(xz + cz[r] + brz);
                float rg = sigmoidf_(xr + cr[r] + brr);
                float hc = tanhf_(xh + rg * (ch[r] + brh));
                h[db][r] = z * h[db][r] + (1.f - z) * hc;
            }
        }

        // write new h to LDS stage (for next A) and stream to HS (f16)
#pragma unroll
        for (int db = 0; db < 8; ++db)
#pragma unroll
            for (int r = 0; r < 8; ++r) {
                int m = hsel * 8 + r, c = db * 16 + ncol;
                _Float16 v = (_Float16)h[db][r];
                hst[m * 128 + c] = v;
                HS[((size_t)(s0 + m) * L_ + t) * 128 + c] = v;
            }
        __syncthreads();
    }
}

// ----------------------------- attention:  u = last@T, w = sum att*hs, ctx = w@wv
__global__ void k_attn(const _Float16* HS, const _Float16* Tt, const _Float16* WvT,
                       float* out) {
    __shared__ float    ust[4][16 * 128];   // 32 KB
    __shared__ _Float16 wst[4][16 * 128];   // 16 KB
    int widx = threadIdx.x >> 5;
    int lane = threadIdx.x & 31;
    int tile = blockIdx.x * 4 + widx;
    int s0 = tile * 16;
    int hsel = lane >> 4, ncol = lane & 15;

    // u[s,:] = last[s,:] @ T     (last = HS[:, 15, :])
    {
        const _Float16* lrow = HS + ((size_t)(s0 + (lane & 15)) * L_ + 15) * 128;
        v16h af[4];
#pragma unroll
        for (int kc = 0; kc < 4; ++kc) af[kc] = load_afrag(lrow, kc, hsel);
#pragma unroll
        for (int nb = 0; nb < 8; ++nb) {
            v8f acc = {};
#pragma unroll
            for (int kc = 0; kc < 4; ++kc) {
                v16h b = *(const v16h*)(Tt + (nb * 16 + ncol) * 128 + kc * 32 + hsel * 16);
                acc = wmma_(af[kc], b, acc);
            }
#pragma unroll
            for (int r = 0; r < 8; ++r)
                ust[widx][(hsel * 8 + r) * 128 + nb * 16 + ncol] = acc[r];
        }
    }
    __syncthreads();

    // w[s,:] = sum_t (hs_t . u_s) * hs_t ;  lane handles (seq = lane&15, half jh)
    int sl = lane & 15, jh = hsel;
    float ur[64];
#pragma unroll
    for (int i = 0; i < 64; ++i) ur[i] = ust[widx][sl * 128 + jh * 64 + i];
    float w[64];
#pragma unroll
    for (int i = 0; i < 64; ++i) w[i] = 0.f;

    for (int t = 0; t < L_; ++t) {
        const _Float16* hr = HS + ((size_t)(s0 + sl) * L_ + t) * 128 + jh * 64;
        __builtin_prefetch((const void*)(hr + 128), 0, 0);   // next-t row
        float part = 0.f;
#pragma unroll
        for (int i = 0; i < 64; ++i) part += (float)hr[i] * ur[i];
        float att = part + __shfl_xor(part, 16, 32);   // both halves get full dot
#pragma unroll
        for (int i = 0; i < 64; ++i) w[i] += att * (float)hr[i];
    }
#pragma unroll
    for (int i = 0; i < 64; ++i)
        wst[widx][sl * 128 + jh * 64 + i] = (_Float16)w[i];
    __syncthreads();

    // ctx = w @ wv  -> out
    {
        v16h af[4];
#pragma unroll
        for (int kc = 0; kc < 4; ++kc)
            af[kc] = load_afrag(&wst[widx][(lane & 15) * 128], kc, hsel);
#pragma unroll
        for (int nb = 0; nb < 8; ++nb) {
            v8f acc = {};
#pragma unroll
            for (int kc = 0; kc < 4; ++kc) {
                v16h b = *(const v16h*)(WvT + (nb * 16 + ncol) * 128 + kc * 32 + hsel * 16);
                acc = wmma_(af[kc], b, acc);
            }
#pragma unroll
            for (int r = 0; r < 8; ++r)
                out[(size_t)(s0 + hsel * 8 + r) * 128 + nb * 16 + ncol] = acc[r];
        }
    }
}

// ---------------------------------------------------------------- launcher
extern "C" void kernel_launch(void* const* d_in, const int* in_sizes, int n_in,
                              void* d_out, int out_size, void* d_ws, size_t ws_size,
                              hipStream_t stream) {
    const float* inputs = (const float*)d_in[0];
    const float* W_in   = (const float*)d_in[1];
    const float* W_rec  = (const float*)d_in[2];
    const float* b_in   = (const float*)d_in[3];
    const float* b_rec  = (const float*)d_in[4];
    const float* wq     = (const float*)d_in[5];
    const float* wk     = (const float*)d_in[6];
    const float* wv     = (const float*)d_in[7];
    const int*   paths  = (const int*)d_in[8];
    const int*   idx    = (const int*)d_in[9];
    const int*   seqs   = (const int*)d_in[10];
    const int    ntot   = in_sizes[8];

    char* ws = (char*)d_ws;
    _Float16* HS    = (_Float16*)(ws);                       // 32768*16*128*2 = 128 MB
    float*    G     = (float*)(ws + 134217728);              // 16384*384*4    = 24 MB
    _Float16* WinT  = (_Float16*)(ws + 159383552);           // 96 KB
    _Float16* WrecT = (_Float16*)(ws + 159481856);           // 96 KB
    _Float16* WvT   = (_Float16*)(ws + 159580160);           // 32 KB
    _Float16* Tt    = (_Float16*)(ws + 159612928);           // 32 KB
    int*      elemtab = (int*)(ws + 159645696);              // 128 KB

    k_init_elem<<<(P_ * L_ + 255) / 256, 256, 0, stream>>>(elemtab);
    k_scatter<<<(ntot + 255) / 256, 256, 0, stream>>>(paths, idx, seqs, ntot, elemtab);
    k_prep<<<(TD_ * DIN_ + 255) / 256, 256, 0, stream>>>(W_in, W_rec, wv, WinT, WrecT, WvT);
    k_T<<<(D_ * D_ + 255) / 256, 256, 0, stream>>>(wq, wk, Tt);

    // G: 1024 row-tiles x 6 col-groups = 6144 waves, 4 waves/block
    k_G<<<1536, 128, 0, stream>>>(inputs, WinT, b_in, G);

    // GRU: 2048 tiles / 4 waves per block, 112 KB dynamic LDS
    static const int kGruLds = 98304 + 4 * 4096;
    hipFuncSetAttribute((const void*)k_gru,
                        hipFuncAttributeMaxDynamicSharedMemorySize, kGruLds);
    k_gru<<<512, 128, kGruLds, stream>>>(G, WrecT, b_in, b_rec, elemtab, HS);

    k_attn<<<512, 128, 0, stream>>>(HS, Tt, WvT, (float*)d_out);
}
